// SelfAttention2D_1846835937365
// MI455X (gfx1250) — matrix operand order repaired and checked by reading, not compiled, not run on hardware
//
#include <hip/hip_runtime.h>
#include <hip/hip_bf16.h>

// ---------------------------------------------------------------------------
// BoTNet-style 2D self-attention for MI455X (gfx1250, wave32, WMMA).
// All GEMM stages use v_wmma_f32_16x16x32_bf16 (f32 accumulate).
// Flash-style attention: the 536MB logits tensor never touches memory;
// working set (~100MB) fits the 192MB L2.
// ---------------------------------------------------------------------------

typedef __bf16 bf16_t;
typedef __attribute__((ext_vector_type(16))) __bf16 v16bf;
typedef __attribute__((ext_vector_type(8)))  float  v8f;

union bfvec16 { uint4 u[2]; v16bf v; };

#define B_   16
#define H_   32
#define W_   32
#define D_   512
#define DK_  256
#define DV_  256
#define NH_  8
#define DKH_ 32
#define DVH_ 32
#define HW_  1024
#define NKQV 768
#define QSCALE 0.17677669529663687f  /* 32^-0.5 */
#define BN_EPS 1e-3f

__device__ __forceinline__ v8f wmma_bf16(v16bf a, v16bf b, v8f c) {
    return __builtin_amdgcn_wmma_f32_16x16x32_bf16(
        /*neg_a=*/false, a, /*neg_b=*/false, b,
        /*c_mod=*/(short)0, c, /*reuse_a=*/false, /*reuse_b=*/false);
}

__device__ __forceinline__ unsigned pack_bf16(float a, float b) {
    union { bf16_t h[2]; unsigned u; } t;
    t.h[0] = (bf16_t)a;
    t.h[1] = (bf16_t)b;
    return t.u;
}

__device__ __forceinline__ uint4 pack8_bf16(float4 a, float4 b) {
    uint4 r;
    r.x = pack_bf16(a.x, a.y);
    r.y = pack_bf16(a.z, a.w);
    r.z = pack_bf16(b.x, b.y);
    r.w = pack_bf16(b.z, b.w);
    return r;
}

// ===========================================================================
// Kernel 1: kqv = x[16384,512] @ w_kqv[512,768] + b_kqv, scattered into
// head-major bf16 buffers: K (j-major,d-contig), Q (pre-scaled), V^T (d-major).
// Block tile 128x128, 8 waves (4 along M x 2 along N), K-step 32.
// LDS tiles stored bf16 with 40-element (80B) row stride -> b128 DS traffic.
// ===========================================================================
#define LDS_STR 40

__global__ __launch_bounds__(256) void kqv_gemm_kernel(
    const float* __restrict__ x, const float* __restrict__ w,
    const float* __restrict__ bias,
    bf16_t* __restrict__ Qh, bf16_t* __restrict__ Kh, bf16_t* __restrict__ Vt)
{
    __shared__ bf16_t sA[128 * LDS_STR];  // [m][k]
    __shared__ bf16_t sB[128 * LDS_STR];  // [n][k] (transposed)

    const int tid  = threadIdx.x;
    const int lane = tid & 31;
    const int wave = tid >> 5;
    const int bm   = blockIdx.x * 128;
    const int bn   = blockIdx.y * 128;
    const int wm   = (wave & 3) * 32;   // wave tile: 32 rows
    const int wn   = (wave >> 2) * 64;  //            64 cols
    const int lh   = lane >> 4;         // lane half
    const int ll   = lane & 15;

    // staging maps
    const int ar  = tid >> 1;           // A row 0..127
    const int ac0 = (tid & 1) * 16;     // A col group
    const int bnn = tid & 127;          // B column (n) 0..127
    const int bkg = (tid >> 7) * 16;    // B k group 0 or 16

    v8f acc[2][4];
#pragma unroll
    for (int s = 0; s < 2; ++s)
#pragma unroll
        for (int t = 0; t < 4; ++t) acc[s][t] = (v8f){};

    for (int k0 = 0; k0 < D_; k0 += 32) {
        // --- stage A: 128x32 f32 -> bf16, packed b128 stores ---
        {
            const float4* src = (const float4*)(x + (size_t)(bm + ar) * D_ + k0 + ac0);
            float4 f0 = src[0], f1 = src[1], f2 = src[2], f3 = src[3];
            *(uint4*)&sA[ar * LDS_STR + ac0]     = pack8_bf16(f0, f1);
            *(uint4*)&sA[ar * LDS_STR + ac0 + 8] = pack8_bf16(f2, f3);
        }
        // --- stage B transposed: each thread owns one column, 16 k-values ---
        {
            const float* src = w + (size_t)(k0 + bkg) * NKQV + bn + bnn;
            float v0[8], v1[8];
#pragma unroll
            for (int j = 0; j < 8; ++j) v0[j] = src[(size_t)j * NKQV];
#pragma unroll
            for (int j = 0; j < 8; ++j) v1[j] = src[(size_t)(8 + j) * NKQV];
            uint4 p0, p1;
            p0.x = pack_bf16(v0[0], v0[1]); p0.y = pack_bf16(v0[2], v0[3]);
            p0.z = pack_bf16(v0[4], v0[5]); p0.w = pack_bf16(v0[6], v0[7]);
            p1.x = pack_bf16(v1[0], v1[1]); p1.y = pack_bf16(v1[2], v1[3]);
            p1.z = pack_bf16(v1[4], v1[5]); p1.w = pack_bf16(v1[6], v1[7]);
            *(uint4*)&sB[bnn * LDS_STR + bkg]     = p0;
            *(uint4*)&sB[bnn * LDS_STR + bkg + 8] = p1;
        }
        if (k0 + 32 < D_) {  // prefetch next A k-tile -> global_prefetch_b8
            __builtin_prefetch(x + (size_t)(bm + ar) * D_ + k0 + 32 + ac0, 0, 3);
        }
        __syncthreads();

        v16bf afrag[2], bfrag[4];
#pragma unroll
        for (int s = 0; s < 2; ++s) {
            bfvec16 t;
            const bf16_t* p = &sA[(wm + s * 16 + ll) * LDS_STR + 16 * lh];
            t.u[0] = *(const uint4*)p;
            t.u[1] = *(const uint4*)(p + 8);
            afrag[s] = t.v;
        }
#pragma unroll
        for (int t4 = 0; t4 < 4; ++t4) {
            bfvec16 t;
            const bf16_t* p = &sB[(wn + t4 * 16 + ll) * LDS_STR + 16 * lh];
            t.u[0] = *(const uint4*)p;
            t.u[1] = *(const uint4*)(p + 8);
            bfrag[t4] = t.v;
        }
#pragma unroll
        for (int s = 0; s < 2; ++s)
#pragma unroll
            for (int t4 = 0; t4 < 4; ++t4)
                acc[s][t4] = wmma_bf16(afrag[s], bfrag[t4], acc[s][t4]);
        __syncthreads();
    }

    // epilogue: bias, split k|q|v, scatter to head-major bf16 buffers
#pragma unroll
    for (int s = 0; s < 2; ++s) {
#pragma unroll
        for (int t = 0; t < 4; ++t) {
            const int gc = bn + wn + t * 16 + ll;
            const float bval = bias[gc];
            const int head = (gc & 255) >> 5;
            const int d    = gc & 31;
#pragma unroll
            for (int r = 0; r < 8; ++r) {
                const int m = bm + wm + s * 16 + r + 8 * lh;
                const int b = m >> 10, i = m & (HW_ - 1);
                const float v = acc[s][t][r] + bval;
                const size_t bnh = (size_t)(b * NH_ + head);
                if (gc < DK_) {
                    Kh[(bnh * HW_ + i) * DKH_ + d] = (bf16_t)v;
                } else if (gc < 2 * DK_) {
                    Qh[(bnh * HW_ + i) * DKH_ + d] = (bf16_t)(v * QSCALE);
                } else {
                    Vt[(bnh * DVH_ + d) * HW_ + i] = (bf16_t)v;
                }
            }
        }
    }
}

// ===========================================================================
// Kernel 2: relative-position tables.
//   PW[bnh,i,w2] = q(i) . rel_w[w2-w1+31]    PH[bnh,i,h2] = q(i) . rel_h[h2-h1+31]
// (q already carries the 32^-0.5 scale, matching the reference.)
// ===========================================================================
__global__ __launch_bounds__(256) void relpos_kernel(
    const bf16_t* __restrict__ Qh, const float* __restrict__ rel_w,
    const float* __restrict__ rel_h, float* __restrict__ PW, float* __restrict__ PH)
{
    const int row = blockIdx.x * 4 + (threadIdx.x >> 6);  // (b*NH+nh)*HW + i
    const int sub = threadIdx.x & 63;
    const int i  = row & (HW_ - 1);
    const int w1 = i & 31, h1 = i >> 5;
    const bf16_t* q = Qh + (size_t)row * DKH_;
    if (sub < 32) {
        const float* rk = rel_w + (size_t)(sub - w1 + 31) * DKH_;
        float a = 0.f;
#pragma unroll
        for (int d = 0; d < 32; ++d) a += (float)q[d] * rk[d];
        PW[(size_t)row * 32 + sub] = a;
    } else {
        const int h2 = sub - 32;
        const float* rk = rel_h + (size_t)(h2 - h1 + 31) * DKH_;
        float a = 0.f;
#pragma unroll
        for (int d = 0; d < 32; ++d) a += (float)q[d] * rk[d];
        PH[(size_t)row * 32 + h2] = a;
    }
}

// ===========================================================================
// Kernel 3: flash attention. Block = (b, head, 128-query tile); 8 waves,
// 16 query rows each. Loop over 32-key chunks (== one image row, so h2 is
// uniform per chunk): 2 QK^T WMMAs -> rel add -> online softmax (shfl-xor
// within 16-lane halves) -> P staged to per-wave LDS -> 2 PV WMMAs.
// ===========================================================================
__global__ __launch_bounds__(256) void attention_kernel(
    const bf16_t* __restrict__ Qh, const bf16_t* __restrict__ Kh,
    const bf16_t* __restrict__ Vt, const float* __restrict__ PW,
    const float* __restrict__ PH, bf16_t* __restrict__ Obuf)
{
    __shared__ bf16_t sP[8][16 * LDS_STR];  // per-wave 16x32 P tile, 80B rows

    const int tid  = threadIdx.x;
    const int lane = tid & 31;
    const int wave = tid >> 5;
    const int lh   = lane >> 4;
    const int ll   = lane & 15;
    const int bnh  = blockIdx.x >> 3;       // b*NH + nh
    const int qt   = blockIdx.x & 7;
    const int b    = bnh >> 3, nh = bnh & 7;
    const int i0   = qt * 128 + wave * 16;  // first query row of this wave

    // Q A-fragment (16 rows x 32 d), persistent
    v16bf qfrag;
    {
        bfvec16 t;
        const bf16_t* p = Qh + ((size_t)bnh * HW_ + i0 + ll) * DKH_ + 16 * lh;
        t.u[0] = *(const uint4*)p;
        t.u[1] = *(const uint4*)(p + 8);
        qfrag = t.v;
    }

    float pw0[8], pw1[8], rmax[8], rsum[8];
    v8f o0 = (v8f){}, o1 = (v8f){};
#pragma unroll
    for (int r = 0; r < 8; ++r) {
        const size_t base = ((size_t)bnh * HW_ + i0 + r + 8 * lh) * 32;
        pw0[r] = PW[base + ll];
        pw1[r] = PW[base + 16 + ll];
        rmax[r] = -1e30f;
        rsum[r] = 0.f;
    }

    const bf16_t* Kbase = Kh + (size_t)bnh * HW_ * DKH_;
    const bf16_t* Vbase = Vt + (size_t)bnh * DVH_ * HW_;
    const float*  PHrow = PH + ((size_t)bnh * HW_) * 32;
    bf16_t* myP = &sP[wave][0];

    for (int kc = 0; kc < 32; ++kc) {
        const int j0 = kc * 32;

        // K B-fragments: cols j0..j0+15 and j0+16..j0+31 (d contiguous, b128)
        v16bf kf0, kf1;
        {
            bfvec16 t0, t1;
            const bf16_t* p0 = Kbase + (size_t)(j0 + ll) * DKH_ + 16 * lh;
            const bf16_t* p1 = p0 + 16 * DKH_;
            t0.u[0] = *(const uint4*)p0;
            t0.u[1] = *(const uint4*)(p0 + 8);
            t1.u[0] = *(const uint4*)p1;
            t1.u[1] = *(const uint4*)(p1 + 8);
            kf0 = t0.v; kf1 = t1.v;
        }
        v8f l0 = (v8f){}, l1 = (v8f){};
        l0 = wmma_bf16(qfrag, kf0, l0);
        l1 = wmma_bf16(qfrag, kf1, l1);

        // rel-position add + per-row tile max
        float mx[8];
#pragma unroll
        for (int r = 0; r < 8; ++r) {
            const float ph = PHrow[(size_t)(i0 + r + 8 * lh) * 32 + kc];
            const float a0 = l0[r] + pw0[r] + ph;
            const float a1 = l1[r] + pw1[r] + ph;
            l0[r] = a0; l1[r] = a1;
            mx[r] = fmaxf(a0, a1);
        }
#pragma unroll
        for (int s = 1; s < 16; s <<= 1)
#pragma unroll
            for (int r = 0; r < 8; ++r)
                mx[r] = fmaxf(mx[r], __shfl_xor(mx[r], s, 32));

        // online softmax update; stage P (bf16) into per-wave LDS
#pragma unroll
        for (int r = 0; r < 8; ++r) {
            const float nm   = fmaxf(rmax[r], mx[r]);
            const float corr = __expf(rmax[r] - nm);
            rmax[r] = nm;
            const float p0 = __expf(l0[r] - nm);
            const float p1 = __expf(l1[r] - nm);
            float ps = p0 + p1;
#pragma unroll
            for (int s = 1; s < 16; s <<= 1) ps += __shfl_xor(ps, s, 32);
            rsum[r] = rsum[r] * corr + ps;
            o0[r] *= corr;
            o1[r] *= corr;
            const int mloc = r + 8 * lh;
            myP[mloc * LDS_STR + ll]      = (bf16_t)p0;
            myP[mloc * LDS_STR + 16 + ll] = (bf16_t)p1;
        }
        // same-wave DS ops are in-order: no barrier needed before readback

        // P A-fragment (16 rows x 32 keys), b128 reloads (80B row stride)
        v16bf pfrag;
        {
            bfvec16 t;
            const bf16_t* p = myP + ll * LDS_STR + 16 * lh;
            t.u[0] = *(const uint4*)p;
            t.u[1] = *(const uint4*)(p + 8);
            pfrag = t.v;
        }
        // V B-fragments from V^T (j contiguous, b128)
        v16bf vf0, vf1;
        {
            bfvec16 t0, t1;
            const bf16_t* p0 = Vbase + (size_t)ll * HW_ + j0 + 16 * lh;
            const bf16_t* p1 = p0 + (size_t)16 * HW_;
            t0.u[0] = *(const uint4*)p0;
            t0.u[1] = *(const uint4*)(p0 + 8);
            t1.u[0] = *(const uint4*)p1;
            t1.u[1] = *(const uint4*)(p1 + 8);
            vf0 = t0.v; vf1 = t1.v;
        }
        o0 = wmma_bf16(pfrag, vf0, o0);
        o1 = wmma_bf16(pfrag, vf1, o1);
    }

    // normalize and emit bf16 o in [b, hw, nh*32+d] layout for the proj GEMM
#pragma unroll
    for (int r = 0; r < 8; ++r) {
        const int m = i0 + r + 8 * lh;
        const float inv = 1.f / rsum[r];
        const size_t orow = ((size_t)b * HW_ + m) * DV_ + nh * DVH_;
        Obuf[orow + ll]      = (bf16_t)(o0[r] * inv);
        Obuf[orow + 16 + ll] = (bf16_t)(o1[r] * inv);
    }
}

// ===========================================================================
// Kernel 4: out = o[16384,256] @ w_proj[256,256] + b_proj, fused BatchNorm.
// ===========================================================================
__global__ __launch_bounds__(256) void proj_gemm_kernel(
    const bf16_t* __restrict__ A, const float* __restrict__ w,
    const float* __restrict__ bias, const float* __restrict__ gamma,
    const float* __restrict__ beta, const float* __restrict__ mean,
    const float* __restrict__ var, float* __restrict__ out)
{
    __shared__ bf16_t sA[128 * LDS_STR];
    __shared__ bf16_t sB[128 * LDS_STR];

    const int tid  = threadIdx.x;
    const int lane = tid & 31;
    const int wave = tid >> 5;
    const int bm   = blockIdx.x * 128;
    const int bn   = blockIdx.y * 128;
    const int wm   = (wave & 3) * 32;
    const int wn   = (wave >> 2) * 64;
    const int lh   = lane >> 4;
    const int ll   = lane & 15;

    const int ar  = tid >> 1;
    const int ac0 = (tid & 1) * 16;
    const int bnn = tid & 127;
    const int bkg = (tid >> 7) * 16;

    v8f acc[2][4];
#pragma unroll
    for (int s = 0; s < 2; ++s)
#pragma unroll
        for (int t = 0; t < 4; ++t) acc[s][t] = (v8f){};

    for (int k0 = 0; k0 < DV_; k0 += 32) {
        // A is already bf16: straight b128 copies
        {
            const uint4* src = (const uint4*)(A + (size_t)(bm + ar) * DV_ + k0 + ac0);
            *(uint4*)&sA[ar * LDS_STR + ac0]     = src[0];
            *(uint4*)&sA[ar * LDS_STR + ac0 + 8] = src[1];
        }
        // B transposed staging, packed stores
        {
            const float* src = w + (size_t)(k0 + bkg) * DV_ + bn + bnn;
            float v0[8], v1[8];
#pragma unroll
            for (int j = 0; j < 8; ++j) v0[j] = src[(size_t)j * DV_];
#pragma unroll
            for (int j = 0; j < 8; ++j) v1[j] = src[(size_t)(8 + j) * DV_];
            uint4 p0, p1;
            p0.x = pack_bf16(v0[0], v0[1]); p0.y = pack_bf16(v0[2], v0[3]);
            p0.z = pack_bf16(v0[4], v0[5]); p0.w = pack_bf16(v0[6], v0[7]);
            p1.x = pack_bf16(v1[0], v1[1]); p1.y = pack_bf16(v1[2], v1[3]);
            p1.z = pack_bf16(v1[4], v1[5]); p1.w = pack_bf16(v1[6], v1[7]);
            *(uint4*)&sB[bnn * LDS_STR + bkg]     = p0;
            *(uint4*)&sB[bnn * LDS_STR + bkg + 8] = p1;
        }
        if (k0 + 32 < DV_) {
            __builtin_prefetch(A + (size_t)(bm + ar) * DV_ + k0 + 32 + ac0, 0, 3);
        }
        __syncthreads();

        v16bf afrag[2], bfrag[4];
#pragma unroll
        for (int s = 0; s < 2; ++s) {
            bfvec16 t;
            const bf16_t* p = &sA[(wm + s * 16 + ll) * LDS_STR + 16 * lh];
            t.u[0] = *(const uint4*)p;
            t.u[1] = *(const uint4*)(p + 8);
            afrag[s] = t.v;
        }
#pragma unroll
        for (int t4 = 0; t4 < 4; ++t4) {
            bfvec16 t;
            const bf16_t* p = &sB[(wn + t4 * 16 + ll) * LDS_STR + 16 * lh];
            t.u[0] = *(const uint4*)p;
            t.u[1] = *(const uint4*)(p + 8);
            bfrag[t4] = t.v;
        }
#pragma unroll
        for (int s = 0; s < 2; ++s)
#pragma unroll
            for (int t4 = 0; t4 < 4; ++t4)
                acc[s][t4] = wmma_bf16(afrag[s], bfrag[t4], acc[s][t4]);
        __syncthreads();
    }

#pragma unroll
    for (int s = 0; s < 2; ++s) {
#pragma unroll
        for (int t = 0; t < 4; ++t) {
            const int gc = bn + wn + t * 16 + ll;
            const float bv = bias[gc];
            const float sc = gamma[gc] * rsqrtf(var[gc] + BN_EPS);
            const float sh = beta[gc] - mean[gc] * sc;
#pragma unroll
            for (int r = 0; r < 8; ++r) {
                const int m = bm + wm + s * 16 + r + 8 * lh;
                out[(size_t)m * DV_ + gc] = (acc[s][t][r] + bv) * sc + sh;
            }
        }
    }
}

// ===========================================================================
extern "C" void kernel_launch(void* const* d_in, const int* in_sizes, int n_in,
                              void* d_out, int out_size, void* d_ws, size_t ws_size,
                              hipStream_t stream) {
    (void)in_sizes; (void)n_in; (void)out_size; (void)ws_size;
    const float* x      = (const float*)d_in[0];
    const float* w_kqv  = (const float*)d_in[1];
    const float* b_kqv  = (const float*)d_in[2];
    const float* w_proj = (const float*)d_in[3];
    const float* b_proj = (const float*)d_in[4];
    const float* gamma  = (const float*)d_in[5];
    const float* beta   = (const float*)d_in[6];
    const float* mmean  = (const float*)d_in[7];
    const float* mvar   = (const float*)d_in[8];
    const float* rel_w  = (const float*)d_in[9];
    const float* rel_h  = (const float*)d_in[10];
    float* out = (float*)d_out;

    // workspace partition (64 MB total)
    char* ws = (char*)d_ws;
    bf16_t* Qh = (bf16_t*)(ws);                         //  8 MB: [B*NH, HW, 32] bf16
    bf16_t* Kh = (bf16_t*)(ws + ((size_t)8  << 20));    //  8 MB: [B*NH, HW, 32] bf16
    bf16_t* Vt = (bf16_t*)(ws + ((size_t)16 << 20));    //  8 MB: [B*NH, 32, HW] bf16
    float*  PW = (float*) (ws + ((size_t)24 << 20));    // 16 MB: [B*NH*HW, 32] f32
    float*  PH = (float*) (ws + ((size_t)40 << 20));    // 16 MB: [B*NH*HW, 32] f32
    bf16_t* Ob = (bf16_t*)(ws + ((size_t)56 << 20));    //  8 MB: [B*HW, 256] bf16

    // 1) kqv GEMM + head-major scatter
    kqv_gemm_kernel<<<dim3(128, 6), 256, 0, stream>>>(x, w_kqv, b_kqv, Qh, Kh, Vt);
    // 2) relative-position tables
    relpos_kernel<<<dim3(B_ * NH_ * HW_ / 4), 256, 0, stream>>>(Qh, rel_w, rel_h, PW, PH);
    // 3) flash attention (B*NH * 8 query tiles)
    attention_kernel<<<dim3(B_ * NH_ * 8), 256, 0, stream>>>(Qh, Kh, Vt, PW, PH, Ob);
    // 4) output projection + BatchNorm
    proj_gemm_kernel<<<dim3(128, 2), 256, 0, stream>>>(Ob, w_proj, b_proj, gamma, beta,
                                                       mmean, mvar, out);
}
